// SparseMoEFFN_45689862095239
// MI455X (gfx1250) — matrix operand order repaired and checked by reading, not compiled, not run on hardware
//
#include <hip/hip_runtime.h>
#include <hip/hip_bf16.h>
#include <cstdint>
#include <cstddef>

// ---------------- problem constants ----------------
static constexpr int T_TOK  = 8192;
static constexpr int C_DIM  = 1024;
static constexpr int H_DIM  = 4096;
static constexpr int E_EXP  = 8;      // routed experts
static constexpr int NE     = 9;      // routed + shared
static constexpr float ALPHA_LOSS = 0.01f;

// ---------------- tiling ----------------
static constexpr int NCH1   = 64;            // P1: H-columns per block
static constexpr int PS1    = C_DIM + 8;     // P1 panel/x row stride (ushorts) -> 2064B, 16B aligned
static constexpr int SPLIT1 = 4;             // P1 token-split factor

static constexpr int NC2    = 64;            // P2: C-columns per block
static constexpr int KH     = 2048;          // P2: K-half (of H)
static constexpr int PS2    = KH + 16;       // P2 panel row stride (ushorts) -> 4128B
static constexpr int SPLIT2 = 4;             // P2 token-split factor

// P1 dynamic LDS layout (bytes)
static constexpr size_t P1_TOKS = 0;                          // int[16]
static constexpr size_t P1_WSC  = 64;                         // float[16]
static constexpr size_t P1_ACC  = 128;                        // float[4][2][16][16] K-split exchange
static constexpr size_t P1_PG   = P1_ACC + 4 * 2 * 256 * 4;   // 8320
static constexpr size_t P1_PU   = P1_PG + (size_t)NCH1 * PS1 * 2;   // +132096
static constexpr size_t P1_XS   = P1_PU + (size_t)NCH1 * PS1 * 2;
static constexpr size_t P1_SMEM = P1_XS + (size_t)16 * PS1 * 2;     // 305,536 B

// P2 dynamic LDS layout
static constexpr size_t P2_TOKS = 0;                          // int[64]
static constexpr size_t P2_PAN  = 256;
static constexpr size_t P2_SMEM = P2_PAN + (size_t)NC2 * PS2 * 2;   // 264,448 B

// ---------------- vector types ----------------
typedef __attribute__((ext_vector_type(16))) __bf16          v16bf;
typedef __attribute__((ext_vector_type(16))) unsigned short  v16us;
typedef __attribute__((ext_vector_type(8)))  unsigned short  v8us;
typedef __attribute__((ext_vector_type(8)))  float           v8f;

__device__ __forceinline__ unsigned short f2bf(float f) {
  unsigned int u = __builtin_bit_cast(unsigned int, f);
  u += 0x7FFFu + ((u >> 16) & 1u);   // RNE
  return (unsigned short)(u >> 16);
}

__device__ __forceinline__ v8f wmma_bf16(v16bf a, v16bf b, v8f c) {
  return __builtin_amdgcn_wmma_f32_16x16x32_bf16(false, a, false, b, (short)0, c,
                                                 false, false);
}

// A frag (16x32): lane m=l%16, h=l/16 -> K {h*8..+7} and {16+h*8..+7}
__device__ __forceinline__ v16bf load_a16(const unsigned short* row, int k0, int h) {
  const v8us lo = *(const v8us*)(row + k0 + h * 8);
  const v8us hi = *(const v8us*)(row + k0 + 16 + h * 8);
  v16us t;
#pragma unroll
  for (int j = 0; j < 8; ++j) { t[j] = lo[j]; t[j + 8] = hi[j]; }
  return __builtin_bit_cast(v16bf, t);
}

// B frag (32x16) from transposed panel row: lane n, h -> K {h*8..+7} and {16+h*8..+7}
__device__ __forceinline__ v16bf load_b16(const unsigned short* prow, int k0, int h) {
  const v8us lo = *(const v8us*)(prow + k0 + h * 8);
  const v8us hi = *(const v8us*)(prow + k0 + 16 + h * 8);
  v16us t;
#pragma unroll
  for (int j = 0; j < 8; ++j) { t[j] = lo[j]; t[j + 8] = hi[j]; }
  return __builtin_bit_cast(v16bf, t);
}

__device__ __forceinline__ float silu_f(float g) {
  return g * __builtin_amdgcn_rcpf(1.f + __expf(-g));
}

// ---------------- x -> bf16 ----------------
__global__ void moe_xcvt_k(const float* __restrict__ x, unsigned short* __restrict__ xbf) {
  const int i0 = (blockIdx.x * blockDim.x + threadIdx.x) * 4;
#pragma unroll
  for (int j = 0; j < 4; ++j) xbf[i0 + j] = f2bf(x[i0 + j]);
}

// ---------------- routing / gating ----------------
__global__ void moe_route_k(const float* __restrict__ x,
                            const float* __restrict__ gate_w,
                            const float* __restrict__ shared_scale,
                            int* __restrict__ cnt, int* __restrict__ lists,
                            float* __restrict__ wlist, float* __restrict__ pisum) {
  __shared__ float gw[E_EXP][C_DIM];
  __shared__ float ps[E_EXP];
  const int tid = threadIdx.x;
  for (int i = tid; i < E_EXP * C_DIM; i += blockDim.x)
    gw[i >> 10][i & (C_DIM - 1)] = gate_w[i];
  if (tid < E_EXP) ps[tid] = 0.f;
  __syncthreads();

  const int t = blockIdx.x * blockDim.x + tid;
  const float* xr = x + (size_t)t * C_DIM;
  float l[E_EXP];
#pragma unroll
  for (int e = 0; e < E_EXP; ++e) l[e] = 0.f;
  for (int c = 0; c < C_DIM; ++c) {
    const float xv = xr[c];
#pragma unroll
    for (int e = 0; e < E_EXP; ++e) l[e] = fmaf(xv, gw[e][c], l[e]);
  }
  float m = l[0];
#pragma unroll
  for (int e = 1; e < E_EXP; ++e) m = fmaxf(m, l[e]);
  float s[E_EXP], sum = 0.f;
#pragma unroll
  for (int e = 0; e < E_EXP; ++e) { s[e] = __expf(l[e] - m); sum += s[e]; }
  const float inv = 1.f / sum;
#pragma unroll
  for (int e = 0; e < E_EXP; ++e) { s[e] *= inv; atomicAdd(&ps[e], s[e]); }

  int i1 = 0; float v1 = s[0];
#pragma unroll
  for (int e = 1; e < E_EXP; ++e) if (s[e] > v1) { v1 = s[e]; i1 = e; }
  int i2 = -1; float v2 = -1.f;
#pragma unroll
  for (int e = 0; e < E_EXP; ++e) if (e != i1 && s[e] > v2) { v2 = s[e]; i2 = e; }
  const float wn = 1.f / (v1 + v2);

  int p1 = atomicAdd(&cnt[i1], 1);
  lists[i1 * T_TOK + p1] = t; wlist[i1 * T_TOK + p1] = v1 * wn;
  int p2 = atomicAdd(&cnt[i2], 1);
  lists[i2 * T_TOK + p2] = t; wlist[i2 * T_TOK + p2] = v2 * wn;
  lists[E_EXP * T_TOK + t] = t;
  wlist[E_EXP * T_TOK + t] = shared_scale[0];

  __syncthreads();
  if (tid < E_EXP) atomicAdd(&pisum[tid], ps[tid]);
}

// ---------------- prefix scan for act-segment bases ----------------
__global__ void moe_scan_k(int* __restrict__ cnt, int* __restrict__ base) {
  if (threadIdx.x == 0) {
    cnt[E_EXP] = T_TOK;
    int acc = 0;
#pragma unroll
    for (int e = 0; e < NE; ++e) { base[e] = acc; acc += cnt[e]; }
  }
}

// ---------------- P1: act = silu(x@Wg)*(x@Wu)*comb_w ----------------
// 8 waves = 4 N-tiles x 2 K-halves; each wave does gate AND up (A reused x2).
__global__ __launch_bounds__(256, 1)
void moe_gateup_k(const unsigned short* __restrict__ xbf,
                  const float* __restrict__ wg, const float* __restrict__ wu,
                  const float* __restrict__ swg, const float* __restrict__ swu,
                  const int* __restrict__ lists, const float* __restrict__ wlist,
                  const int* __restrict__ cnt, const int* __restrict__ base,
                  unsigned short* __restrict__ actws) {
  const int chunk = blockIdx.x, e = blockIdx.y, split = blockIdx.z;
  const int n0g = chunk * NCH1;
  const int count = cnt[e];
  const int bas = base[e];
  const float* Wg = (e < E_EXP) ? wg + (size_t)e * C_DIM * H_DIM : swg;
  const float* Wu = (e < E_EXP) ? wu + (size_t)e * C_DIM * H_DIM : swu;

  extern __shared__ unsigned char smem[];
  int*            toks = (int*)(smem + P1_TOKS);
  float*          wsc  = (float*)(smem + P1_WSC);
  float*          accf = (float*)(smem + P1_ACC);
  unsigned short* pg   = (unsigned short*)(smem + P1_PG);
  unsigned short* pu   = (unsigned short*)(smem + P1_PU);
  unsigned short* xs   = (unsigned short*)(smem + P1_XS);

  const int tid = threadIdx.x;
  // stage transposed bf16 panels: panel[n][k] = W[k][n0g+n]
  for (int i = tid; i < 2 * NCH1 * C_DIM; i += 256) {
    const int mat = i >= NCH1 * C_DIM;
    const int j = mat ? i - NCH1 * C_DIM : i;
    const int k = j >> 6, n = j & (NCH1 - 1);
    const float v = (mat ? Wu : Wg)[(size_t)k * H_DIM + n0g + n];
    (mat ? pu : pg)[n * PS1 + k] = f2bf(v);
  }

  const int wave = tid >> 5, lane = tid & 31;
  const int h = lane >> 4, m = lane & 15, nn = lane & 15;
  const int ntile = wave & 3, kh2 = wave >> 2;     // K-half owner
  const int kbeg = kh2 * (C_DIM / 2);
  const unsigned short* pgrow = pg + (size_t)(ntile * 16 + nn) * PS1;
  const unsigned short* purow = pu + (size_t)(ntile * 16 + nn) * PS1;
  const unsigned short* arow  = xs + (size_t)m * PS1;
  float* exg = accf + (size_t)ntile * 512;
  float* exu = exg + 256;

  const int mtiles = (count + 15) >> 4;
  for (int mt = split; mt < mtiles; mt += SPLIT1) {
    __syncthreads();                               // reuse of toks/xs/accf
    if (tid < 16) {
      const int idx = mt * 16 + tid;
      if (idx < count) { toks[tid] = lists[e * T_TOK + idx]; wsc[tid] = wlist[e * T_TOK + idx]; }
      else             { toks[tid] = 0;                       wsc[tid] = 0.f; }
    }
    __syncthreads();

    // stage 16 bf16 token rows via CDNA5 async global->LDS copies (16B/lane-op)
    for (int i = tid; i < 16 * (C_DIM / 8); i += 256) {      // 2048 x 16B
      const int r = i >> 7, c16 = i & 127;
      const unsigned int ldsa =
          (unsigned int)(uintptr_t)(const void*)(xs + (size_t)r * PS1 + c16 * 8);
      const unsigned short* src = xbf + (size_t)toks[r] * C_DIM + c16 * 8;
      asm volatile("global_load_async_to_lds_b128 %0, %1, off"
                   :: "v"(ldsa), "v"(src) : "memory");
    }
    asm volatile("s_wait_asynccnt 0x0" ::: "memory");
    __syncthreads();

    v8f accg = {}, accu = {};
#pragma unroll 2
    for (int kk = 0; kk < C_DIM / 2; kk += 32) {
      const int k0 = kbeg + kk;
      const v16bf a = load_a16(arow, k0, h);
      accg = wmma_bf16(a, load_b16(pgrow, k0, h), accg);
      accu = wmma_bf16(a, load_b16(purow, k0, h), accu);
    }

    if (kh2 == 1) {                                // publish partial sums
#pragma unroll
      for (int r = 0; r < 8; ++r) {
        exg[(r + h * 8) * 16 + nn] = accg[r];
        exu[(r + h * 8) * 16 + nn] = accu[r];
      }
    }
    __syncthreads();
    if (kh2 == 0) {                                // reduce + silu + emit
#pragma unroll
      for (int r = 0; r < 8; ++r) {
        const int row = r + h * 8;
        if (mt * 16 + row < count) {
          const float g = accg[r] + exg[row * 16 + nn];
          const float u = accu[r] + exu[row * 16 + nn];
          const float a = silu_f(g) * u * wsc[row];
          actws[(size_t)(bas + mt * 16 + row) * H_DIM + n0g + ntile * 16 + nn] = f2bf(a);
        }
      }
    }
  }
}

// ---------------- P2: y[tok] += act @ Wd ----------------
// 8 waves = 4 M-tiles x 2 N-tile-pairs; A reused across 2 WMMAs.
__global__ __launch_bounds__(256, 1)
void moe_down_k(const unsigned short* __restrict__ actws,
                const float* __restrict__ wd, const float* __restrict__ swd,
                const int* __restrict__ lists, const int* __restrict__ cnt,
                const int* __restrict__ base, float* __restrict__ y) {
  const int cchunk = blockIdx.x;
  const int kh = blockIdx.y >> 2, split = blockIdx.y & (SPLIT2 - 1);
  const int e = blockIdx.z;
  const int count = cnt[e];
  const int bas = base[e];
  const float* Wd = (e < E_EXP) ? wd + (size_t)e * H_DIM * C_DIM : swd;
  const int n0 = cchunk * NC2;
  const int kbase = kh * KH;

  extern __shared__ unsigned char smem[];
  int*            toks  = (int*)(smem + P2_TOKS);
  unsigned short* panel = (unsigned short*)(smem + P2_PAN);

  const int tid = threadIdx.x;
  for (int i = tid; i < KH * NC2; i += 256) {      // panel[n][k] = Wd[kbase+k][n0+n]
    const int k = i >> 6, n = i & (NC2 - 1);
    panel[n * PS2 + k] = f2bf(Wd[(size_t)(kbase + k) * C_DIM + n0 + n]);
  }

  const int wave = tid >> 5, lane = tid & 31;
  const int h = lane >> 4, m = lane & 15, nn = lane & 15;
  const int mt = wave >> 1, npair = wave & 1;
  const unsigned short* prow0 = panel + (size_t)((npair * 2 + 0) * 16 + nn) * PS2;
  const unsigned short* prow1 = panel + (size_t)((npair * 2 + 1) * 16 + nn) * PS2;

  const int iters = (count + 63) >> 6;
  for (int it = split; it < iters; it += SPLIT2) {
    __syncthreads();
    if (tid < 64) {
      const int idx = it * 64 + tid;
      toks[tid] = (idx < count) ? lists[e * T_TOK + idx] : 0;
    }
    __syncthreads();

    const unsigned short* arow =
        actws + (size_t)(bas + it * 64 + mt * 16 + m) * H_DIM + kbase;
    v8f acc0 = {}, acc1 = {};
#pragma unroll 2
    for (int k0 = 0; k0 < KH; k0 += 32) {
      const v16bf a = load_a16(arow, k0, h);
      acc0 = wmma_bf16(a, load_b16(prow0, k0, h), acc0);
      acc1 = wmma_bf16(a, load_b16(prow1, k0, h), acc1);
    }
#pragma unroll
    for (int r = 0; r < 8; ++r) {
      const int row = mt * 16 + r + h * 8;
      if (it * 64 + row < count) {
        float* yr = y + (size_t)toks[row] * C_DIM + n0;
        atomicAdd(yr + (npair * 2 + 0) * 16 + nn, acc0[r]);
        atomicAdd(yr + (npair * 2 + 1) * 16 + nn, acc1[r]);
      }
    }
  }
}

// ---------------- aux loss ----------------
__global__ void moe_aux_k(const int* __restrict__ cnt, const float* __restrict__ pisum,
                          float* __restrict__ out_aux) {
  if (threadIdx.x == 0) {
    float s = 0.f;
#pragma unroll
    for (int e = 0; e < E_EXP; ++e) {
      const float Pi = pisum[e] / (float)T_TOK;
      const float fi = ((float)cnt[e] / (float)(T_TOK * 2)) * (float)E_EXP;
      s += Pi * fi;
    }
    *out_aux = s * ALPHA_LOSS;
  }
}

// ---------------- launcher ----------------
extern "C" void kernel_launch(void* const* d_in, const int* in_sizes, int n_in,
                              void* d_out, int out_size, void* d_ws, size_t ws_size,
                              hipStream_t stream) {
  (void)in_sizes; (void)n_in; (void)ws_size;
  const float* x      = (const float*)d_in[0];
  const float* gate_w = (const float*)d_in[1];
  const float* wg     = (const float*)d_in[2];
  const float* wu     = (const float*)d_in[3];
  const float* wd     = (const float*)d_in[4];
  const float* swg    = (const float*)d_in[5];
  const float* swu    = (const float*)d_in[6];
  const float* swd    = (const float*)d_in[7];
  const float* sscale = (const float*)d_in[8];
  float* y = (float*)d_out;

  char* ws = (char*)d_ws;
  int*   cnt   = (int*)ws;                          // [16]
  float* pisum = (float*)(ws + 64);                 // [8]
  int*   base  = (int*)(ws + 128);                  // [16]
  int*   lists = (int*)(ws + 256);                  // [NE*T]
  float* wlist = (float*)(ws + 256 + (size_t)NE * T_TOK * 4);
  unsigned short* xbf = (unsigned short*)(ws + 590080);      // [T*C] bf16
  unsigned short* act = (unsigned short*)(ws + 17367296);    // [(3T+64)*H] bf16

  hipMemsetAsync(d_ws, 0, 256, stream);
  hipMemsetAsync(d_out, 0, (size_t)out_size * sizeof(float), stream);

  moe_xcvt_k<<<(T_TOK * C_DIM) / (256 * 4), 256, 0, stream>>>(x, xbf);
  moe_route_k<<<T_TOK / 256, 256, 0, stream>>>(x, gate_w, sscale, cnt, lists, wlist, pisum);
  moe_scan_k<<<1, 32, 0, stream>>>(cnt, base);

  hipFuncSetAttribute(reinterpret_cast<const void*>(moe_gateup_k),
                      hipFuncAttributeMaxDynamicSharedMemorySize, (int)P1_SMEM);
  moe_gateup_k<<<dim3(H_DIM / NCH1, NE, SPLIT1), 256, P1_SMEM, stream>>>(
      xbf, wg, wu, swg, swu, lists, wlist, cnt, base, act);

  hipFuncSetAttribute(reinterpret_cast<const void*>(moe_down_k),
                      hipFuncAttributeMaxDynamicSharedMemorySize, (int)P2_SMEM);
  moe_down_k<<<dim3(C_DIM / NC2, 2 * SPLIT2, NE), 256, P2_SMEM, stream>>>(
      act, wd, swd, lists, cnt, base, y);

  moe_aux_k<<<1, 32, 0, stream>>>(cnt, pisum, y + (size_t)T_TOK * C_DIM);
}
